// HLayer_16587163697553
// MI455X (gfx1250) — matrix operand order repaired and checked by reading, not compiled, not run on hardware
//
#include <hip/hip_runtime.h>
#include <hip/hip_bf16.h>
#include <math.h>
#include <stdint.h>

typedef __attribute__((ext_vector_type(16))) __bf16 bfvec16;
typedef __attribute__((ext_vector_type(8)))  __bf16 bfvec8;
typedef __attribute__((ext_vector_type(8)))  float  fvec8;

#define N_USER 50000
#define N_ITEM 50000
#define NEDGE  400000

// ---------------------------------------------------------------------------
// GEMM: C[M,128] = A[M,128] @ W[128,128] (+bias)(opt tanh), A in bf16 row-major,
// W pre-transposed bf16 Wt[n][k]. Block = 256 thr = 8 waves, 64 rows/block.
// Staging is a pure 16B copy -> global_load_async_to_lds_b128 (ASYNCcnt),
// then each wave runs 4 sub-tiles x 4 chained v_wmma_f32_16x16x32_bf16.
// ---------------------------------------------------------------------------
__global__ __launch_bounds__(256)
void gemm_k128(const __bf16* __restrict__ A, const __bf16* __restrict__ Wt,
               const float* __restrict__ bias, float* __restrict__ C,
               __bf16* __restrict__ Cbf, int M, int act)
{
    __shared__ __align__(16) __bf16 As[64][136];   // pitch 272B = 16B-aligned
    __shared__ __align__(16) __bf16 Bs[128][136];

    const int t    = threadIdx.x;
    const int row0 = blockIdx.x * 64;

    // Async-stage A strip: 64x128 bf16 = 1024 x 16B chunks.
    for (int i = t; i < 1024; i += 256) {
        int r = i >> 4, c8 = (i & 15) << 3;
        if (row0 + r < M) {
            unsigned lds = (unsigned)(uintptr_t)&As[r][c8];
            unsigned long long g =
                (unsigned long long)(uintptr_t)(A + (size_t)(row0 + r) * 128 + c8);
            asm volatile("global_load_async_to_lds_b128 %0, %1, off"
                         :: "v"(lds), "v"(g) : "memory");
        }
    }
    // Async-stage Wt: 128x128 bf16 = 2048 x 16B chunks.
    for (int i = t; i < 2048; i += 256) {
        int r = i >> 4, c8 = (i & 15) << 3;
        unsigned lds = (unsigned)(uintptr_t)&Bs[r][c8];
        unsigned long long g =
            (unsigned long long)(uintptr_t)(Wt + (size_t)r * 128 + c8);
        asm volatile("global_load_async_to_lds_b128 %0, %1, off"
                     :: "v"(lds), "v"(g) : "memory");
    }
    asm volatile("s_wait_asynccnt 0x0" ::: "memory");
    __syncthreads();

    const int wave = t >> 5;
    const int lane = t & 31;
    const int n0   = wave << 4;
    const int mrow = lane & 15;        // A: row within sub-tile
    const int kg   = lane >> 4;        // K-group per ISA fragment layout
    const int ncol = n0 + (lane & 15); // B/C/D: column

    fvec8 acc[4] = {{}, {}, {}, {}};
    #pragma unroll
    for (int kk = 0; kk < 4; ++kk) {
        const int K0 = kk * 32;
        // B fragment: K = {K0+kg*8..+7} and {K0+16+kg*8..+7} of column ncol
        bfvec8 blo = *reinterpret_cast<const bfvec8*>(&Bs[ncol][K0 + kg * 8]);
        bfvec8 bhi = *reinterpret_cast<const bfvec8*>(&Bs[ncol][K0 + 16 + kg * 8]);
        bfvec16 b;
        #pragma unroll
        for (int j = 0; j < 8; ++j) { b[j] = blo[j]; b[j + 8] = bhi[j]; }
        #pragma unroll
        for (int s = 0; s < 4; ++s) {
            // A fragment: K = {K0+kg*8..+7} and {K0+16+kg*8..+7}
            bfvec8 alo = *reinterpret_cast<const bfvec8*>(&As[s * 16 + mrow][K0 + kg * 8]);
            bfvec8 ahi = *reinterpret_cast<const bfvec8*>(&As[s * 16 + mrow][K0 + 16 + kg * 8]);
            bfvec16 a;
            #pragma unroll
            for (int j = 0; j < 8; ++j) { a[j] = alo[j]; a[j + 8] = ahi[j]; }
            acc[s] = __builtin_amdgcn_wmma_f32_16x16x32_bf16(
                false, a, false, b, (short)0, acc[s], false, false);
        }
    }

    const float bv = bias ? bias[ncol] : 0.0f;
    #pragma unroll
    for (int s = 0; s < 4; ++s) {
        int rbase = row0 + s * 16;
        if (rbase < M) {               // uniform; M % 16 == 0
            #pragma unroll
            for (int r = 0; r < 8; ++r) {
                int row = rbase + r + ((lane >> 4) << 3);
                float val = acc[s][r] + bv;
                if (act == 1) val = tanhf(val);
                size_t idx = (size_t)row * 128 + ncol;
                C[idx] = val;
                if (Cbf) Cbf[idx] = (__bf16)val;
            }
        }
    }
}

// ---------------------------------------------------------------------------
// Small conversion / setup kernels
// ---------------------------------------------------------------------------
__global__ void cvt_bf16(const float* __restrict__ in, __bf16* __restrict__ out, int n)
{
    int tid = blockIdx.x * blockDim.x + threadIdx.x;
    if (tid < n) out[tid] = (__bf16)in[tid];
}

// W[128,128] f32 -> transposed bf16 Wt[n][k]
__global__ void cvt_wt(const float* __restrict__ W, __bf16* __restrict__ Wt)
{
    int i = blockIdx.x * blockDim.x + threadIdx.x;
    if (i < 16384) { int k = i >> 7, n = i & 127; Wt[n * 128 + k] = (__bf16)W[k * 128 + n]; }
}

// rel[8][16][16] -> transposed bf16 block-diagonal Wt[n][k]
__global__ void expand_rel_t(const float* __restrict__ rel, __bf16* __restrict__ out)
{
    for (int i = threadIdx.x + blockIdx.x * blockDim.x; i < 128 * 128;
         i += blockDim.x * gridDim.x) {
        int n = i >> 7, k = i & 127;
        int hn = n >> 4, hk = k >> 4;
        out[i] = (hn == hk) ? (__bf16)rel[(hn << 8) + ((k & 15) << 4) + (n & 15)]
                            : (__bf16)0.0f;
    }
}

// b'[h*16+j] = sum_i b[h*16+i] * rel[h][i][j]
__global__ void compose_bias(const float* __restrict__ b,
                             const float* __restrict__ rel,
                             float* __restrict__ out)
{
    int c = threadIdx.x;
    if (c >= 128) return;
    int h = c >> 4, j = c & 15;
    float s = 0.f;
    #pragma unroll
    for (int i = 0; i < 16; ++i) s += b[(h << 4) + i] * rel[(h << 8) + (i << 4) + j];
    out[c] = s;
}

// ---------------------------------------------------------------------------
// Edge phase
// ---------------------------------------------------------------------------
__global__ __launch_bounds__(256)
void edge_scores(const float* __restrict__ q, const float* __restrict__ k,
                 const int* __restrict__ src, const int* __restrict__ dst,
                 const float* __restrict__ rp, float* __restrict__ raw,
                 float* __restrict__ rawinv, int EH)
{
    int tid = blockIdx.x * blockDim.x + threadIdx.x;
    if (tid >= EH) return;
    int e = tid >> 3, h = tid & 7;
    int s = src[e], d = dst[e];
    const float4* qr = reinterpret_cast<const float4*>(q + (size_t)d * 128 + h * 16);
    const float4* kr = reinterpret_cast<const float4*>(k + (size_t)s * 128 + h * 16);
    float acc = 0.f;
    #pragma unroll
    for (int i = 0; i < 4; ++i) {
        float4 a = qr[i], b = kr[i];
        acc += a.x * b.x + a.y * b.y + a.z * b.z + a.w * b.w;
    }
    float v = acc * rp[h] * 0.25f;
    raw[tid]    = v;
    rawinv[tid] = 1.0f / (v + 1e-6f);
}

__device__ __forceinline__ unsigned f2o(float f)
{
    unsigned u = __float_as_uint(f);
    return (u & 0x80000000u) ? ~u : (u | 0x80000000u);
}
__device__ __forceinline__ float o2f(unsigned u)
{
    return __uint_as_float((u & 0x80000000u) ? (u & 0x7fffffffu) : ~u);
}

__global__ __launch_bounds__(256)
void seg_max(const float* __restrict__ raw, const float* __restrict__ rawinv,
             const int* __restrict__ dst, unsigned* __restrict__ mlf,
             unsigned* __restrict__ mhf, int EH)
{
    int tid = blockIdx.x * blockDim.x + threadIdx.x;
    if (tid >= EH) return;
    int e = tid >> 3, h = tid & 7;
    int idx = dst[e] * 8 + h;
    atomicMax(&mlf[idx], f2o(raw[tid]));
    atomicMax(&mhf[idx], f2o(rawinv[tid]));
}

__global__ __launch_bounds__(256)
void exp_sum(float* __restrict__ raw, float* __restrict__ rawinv,
             const int* __restrict__ dst, const unsigned* __restrict__ mlf,
             const unsigned* __restrict__ mhf, float* __restrict__ zlf,
             float* __restrict__ zhf, int EH)
{
    int tid = blockIdx.x * blockDim.x + threadIdx.x;
    if (tid >= EH) return;
    int e = tid >> 3, h = tid & 7;
    int idx = dst[e] * 8 + h;
    float e1 = __expf(raw[tid] - o2f(mlf[idx]));
    raw[tid] = e1;
    atomicAdd(&zlf[idx], e1);
    float e2 = __expf(rawinv[tid] - o2f(mhf[idx]));
    rawinv[tid] = e2;
    atomicAdd(&zhf[idx], e2);
}

__global__ __launch_bounds__(256)
void aggregate(const float* __restrict__ v, const int* __restrict__ src,
               const int* __restrict__ dst, const float* __restrict__ elf,
               const float* __restrict__ ehf, const float* __restrict__ zlf,
               const float* __restrict__ zhf, float* __restrict__ lt,
               float* __restrict__ ht, int E)
{
    long long tid = (long long)blockIdx.x * blockDim.x + threadIdx.x;
    if (tid >= (long long)E * 32) return;
    int e = (int)(tid >> 5), lane = (int)(tid & 31);
    int c0 = lane << 2, h = c0 >> 4;
    int s = src[e], d = dst[e];
    int idx = d * 8 + h;
    float wl = elf[e * 8 + h] / zlf[idx];
    float wh = ehf[e * 8 + h] / zhf[idx];
    float4 vv = *reinterpret_cast<const float4*>(v + (size_t)s * 128 + c0);
    float* lp = lt + (size_t)d * 128 + c0;
    float* hp = ht + (size_t)d * 128 + c0;
    atomicAdd(lp + 0, vv.x * wl); atomicAdd(lp + 1, vv.y * wl);
    atomicAdd(lp + 2, vv.z * wl); atomicAdd(lp + 3, vv.w * wl);
    atomicAdd(hp + 0, vv.x * wh); atomicAdd(hp + 1, vv.y * wh);
    atomicAdd(hp + 2, vv.z * wh); atomicAdd(hp + 3, vv.w * wh);
}

// ---------------------------------------------------------------------------
// Update phase
// ---------------------------------------------------------------------------
__global__ void prep_ltht(const float* __restrict__ ori, float* __restrict__ lt,
                          float* __restrict__ ht, __bf16* __restrict__ ltb,
                          __bf16* __restrict__ htb, int n)
{
    int tid = blockIdx.x * blockDim.x + threadIdx.x;
    if (tid >= n) return;
    float o = ori[tid];
    float L = o + lt[tid];
    float H = o - ht[tid];
    lt[tid] = L; ht[tid] = H;
    ltb[tid] = (__bf16)L; htb[tid] = (__bf16)H;
}

__global__ __launch_bounds__(256)
void mix_kernel(const float* __restrict__ g0, const float* __restrict__ g1,
                const float* __restrict__ xp, const float* __restrict__ lt,
                const float* __restrict__ ht, float* __restrict__ t,
                __bf16* __restrict__ tb, int n)
{
    int node = blockIdx.x * 8 + (threadIdx.x >> 5);
    int lane = threadIdx.x & 31;
    if (node >= n) return;
    size_t base = (size_t)node * 128 + lane * 4;
    float4 x = *reinterpret_cast<const float4*>(xp + base);
    float4 a = *reinterpret_cast<const float4*>(g0 + base);
    float4 b = *reinterpret_cast<const float4*>(g1 + base);
    float s0 = a.x * x.x + a.y * x.y + a.z * x.z + a.w * x.w;
    float s1 = b.x * x.x + b.y * x.y + b.z * x.z + b.w * x.w;
    #pragma unroll
    for (int off = 16; off > 0; off >>= 1) {
        s0 += __shfl_xor(s0, off, 32);
        s1 += __shfl_xor(s1, off, 32);
    }
    float m = fmaxf(s0, s1);
    float e0 = __expf(s0 - m), e1 = __expf(s1 - m);
    float inv = 1.0f / (e0 + e1);
    float c0 = e0 * inv, c1 = e1 * inv;
    float4 L = *reinterpret_cast<const float4*>(lt + base);
    float4 H = *reinterpret_cast<const float4*>(ht + base);
    float4 o;
    o.x = L.x * c0 + H.x * c1; o.y = L.y * c0 + H.y * c1;
    o.z = L.z * c0 + H.z * c1; o.w = L.w * c0 + H.w * c1;
    *reinterpret_cast<float4*>(t + base) = o;
    tb[base + 0] = (__bf16)o.x; tb[base + 1] = (__bf16)o.y;
    tb[base + 2] = (__bf16)o.z; tb[base + 3] = (__bf16)o.w;
}

__global__ void blend(const float* __restrict__ u, const float* __restrict__ h,
                      const float* __restrict__ skip, int sidx,
                      float* __restrict__ out, int n)
{
    int tid = blockIdx.x * blockDim.x + threadIdx.x;
    if (tid >= n) return;
    float a = 1.0f / (1.0f + __expf(-skip[sidx]));
    out[tid] = u[tid] * a + h[tid] * (1.0f - a);
}

// ---------------------------------------------------------------------------
extern "C" void kernel_launch(void* const* d_in, const int* in_sizes, int n_in,
                              void* d_out, int out_size, void* d_ws, size_t ws_size,
                              hipStream_t stream)
{
    const float* h_user  = (const float*)d_in[0];
    const float* h_item  = (const float*)d_in[1];
    const int*   buys_src = (const int*)d_in[2];
    const int*   buys_dst = (const int*)d_in[3];
    const int*   rev_src  = (const int*)d_in[4];
    const int*   rev_dst  = (const int*)d_in[5];
    const float* Wk_u = (const float*)d_in[6];  const float* bk_u = (const float*)d_in[7];
    const float* Wk_i = (const float*)d_in[8];  const float* bk_i = (const float*)d_in[9];
    const float* Wq_u = (const float*)d_in[10]; const float* bq_u = (const float*)d_in[11];
    const float* Wq_i = (const float*)d_in[12]; const float* bq_i = (const float*)d_in[13];
    const float* Wv_u = (const float*)d_in[14]; const float* bv_u = (const float*)d_in[15];
    const float* Wv_i = (const float*)d_in[16]; const float* bv_i = (const float*)d_in[17];
    const float* Wa_u = (const float*)d_in[18]; const float* ba_u = (const float*)d_in[19];
    const float* Wa_i = (const float*)d_in[20]; const float* ba_i = (const float*)d_in[21];
    const float* Wf   = (const float*)d_in[22]; const float* bF   = (const float*)d_in[23];
    const float* Wx   = (const float*)d_in[24]; const float* bx   = (const float*)d_in[25];
    const float* rel_pri = (const float*)d_in[26];
    const float* rel_att = (const float*)d_in[27];
    const float* rel_msg = (const float*)d_in[28];
    const float* skip    = (const float*)d_in[29];

    float* p = (float*)d_ws;
    size_t o = 0;
    auto take  = [&](size_t n) { n = (n + 7) & ~(size_t)7; float* r = p + o; o += n; return r; };
    auto takeb = [&](size_t n) { return (__bf16*)take((n + 1) / 2); };

    // bf16 weights (transposed Wt[n][k]) and activations
    __bf16* RattT0 = takeb(16384); __bf16* RattT1 = takeb(16384);
    __bf16* RmsgT0 = takeb(16384); __bf16* RmsgT1 = takeb(16384);
    __bf16* WcT0 = takeb(16384);  __bf16* WcT1 = takeb(16384);
    __bf16* WqUT = takeb(16384);  __bf16* WqIT = takeb(16384);
    __bf16* WvUT = takeb(16384);  __bf16* WvIT = takeb(16384);
    __bf16* WfT  = takeb(16384);  __bf16* WxT  = takeb(16384);
    __bf16* WaUT = takeb(16384);  __bf16* WaIT = takeb(16384);
    __bf16* WkUb = takeb(16384);  __bf16* WkIb = takeb(16384);
    __bf16* hUb  = takeb((size_t)N_USER * 128);
    __bf16* hIb  = takeb((size_t)N_ITEM * 128);
    __bf16* oriUb = takeb((size_t)N_USER * 128);
    __bf16* oriIb = takeb((size_t)N_ITEM * 128);
    __bf16* ltUb = takeb((size_t)N_USER * 128); __bf16* htUb = takeb((size_t)N_USER * 128);
    __bf16* ltIb = takeb((size_t)N_ITEM * 128); __bf16* htIb = takeb((size_t)N_ITEM * 128);
    __bf16* tbb  = takeb((size_t)N_USER * 128);

    float* Wc0f = take(16384); float* Wc1f = take(16384);
    float* bc0  = take(128);   float* bc1  = take(128);
    float* k0   = take((size_t)N_USER * 128); float* q0 = take((size_t)N_ITEM * 128);
    float* k1   = take((size_t)N_ITEM * 128); float* q1 = take((size_t)N_USER * 128);
    float* oriU = take((size_t)N_USER * 128); float* oriI = take((size_t)N_ITEM * 128);
    float* v0   = take((size_t)N_USER * 128); float* v1 = take((size_t)N_ITEM * 128);
    float* raw0 = take((size_t)NEDGE * 8); float* rin0 = take((size_t)NEDGE * 8);
    float* raw1 = take((size_t)NEDGE * 8); float* rin1 = take((size_t)NEDGE * 8);
    float* zeroBase = p + o;
    unsigned* mlfI = (unsigned*)take((size_t)N_ITEM * 8);
    unsigned* mhfI = (unsigned*)take((size_t)N_ITEM * 8);
    unsigned* mlfU = (unsigned*)take((size_t)N_USER * 8);
    unsigned* mhfU = (unsigned*)take((size_t)N_USER * 8);
    float* zlfI = take((size_t)N_ITEM * 8); float* zhfI = take((size_t)N_ITEM * 8);
    float* zlfU = take((size_t)N_USER * 8); float* zhfU = take((size_t)N_USER * 8);
    float* ltI = take((size_t)N_ITEM * 128); float* htI = take((size_t)N_ITEM * 128);
    float* ltU = take((size_t)N_USER * 128); float* htU = take((size_t)N_USER * 128);
    size_t zeroBytes = (size_t)((p + o) - zeroBase) * sizeof(float);

    // Update-phase f32 scratch reuses edge-phase buffers.
    float* g0b = k0; float* g1b = q0; float* xpb = k1; float* tb = q1; float* ub = v0;

    float* outU = (float*)d_out;
    float* outI = outU + (size_t)N_USER * 128;

    hipMemsetAsync(zeroBase, 0, zeroBytes, stream);

    // bf16 conversions of GEMM A-operands and weights.
    const int NU128 = N_USER * 128, NI128 = N_ITEM * 128;
    const int BN = (NU128 + 255) / 256, BNI = (NI128 + 255) / 256;
    cvt_bf16<<<BN, 256, 0, stream>>>(h_user, hUb, NU128);
    cvt_bf16<<<BNI, 256, 0, stream>>>(h_item, hIb, NI128);
    cvt_bf16<<<64, 256, 0, stream>>>(Wk_u, WkUb, 16384);
    cvt_bf16<<<64, 256, 0, stream>>>(Wk_i, WkIb, 16384);
    expand_rel_t<<<16, 256, 0, stream>>>(rel_att, RattT0);
    expand_rel_t<<<16, 256, 0, stream>>>(rel_att + 2048, RattT1);
    expand_rel_t<<<16, 256, 0, stream>>>(rel_msg, RmsgT0);
    expand_rel_t<<<16, 256, 0, stream>>>(rel_msg + 2048, RmsgT1);
    cvt_wt<<<64, 256, 0, stream>>>(Wq_u, WqUT); cvt_wt<<<64, 256, 0, stream>>>(Wq_i, WqIT);
    cvt_wt<<<64, 256, 0, stream>>>(Wv_u, WvUT); cvt_wt<<<64, 256, 0, stream>>>(Wv_i, WvIT);
    cvt_wt<<<64, 256, 0, stream>>>(Wf, WfT);    cvt_wt<<<64, 256, 0, stream>>>(Wx, WxT);
    cvt_wt<<<64, 256, 0, stream>>>(Wa_u, WaUT); cvt_wt<<<64, 256, 0, stream>>>(Wa_i, WaIT);
    compose_bias<<<1, 128, 0, stream>>>(bk_u, rel_att, bc0);
    compose_bias<<<1, 128, 0, stream>>>(bk_i, rel_att + 2048, bc1);

    // Compose K-path weights: Wc = Wk @ Ratt', then transpose-convert.
    gemm_k128<<<2, 256, 0, stream>>>(WkUb, RattT0, nullptr, Wc0f, nullptr, 128, 0);
    gemm_k128<<<2, 256, 0, stream>>>(WkIb, RattT1, nullptr, Wc1f, nullptr, 128, 0);
    cvt_wt<<<64, 256, 0, stream>>>(Wc0f, WcT0);
    cvt_wt<<<64, 256, 0, stream>>>(Wc1f, WcT1);

    // Node projections (all WMMA, async-staged).
    const int GU = (N_USER + 63) / 64, GI = (N_ITEM + 63) / 64;
    gemm_k128<<<GU, 256, 0, stream>>>(hUb, WcT0, bc0, k0, nullptr, N_USER, 0);
    gemm_k128<<<GI, 256, 0, stream>>>(hIb, WqIT, bq_i, q0, nullptr, N_ITEM, 0);
    gemm_k128<<<GI, 256, 0, stream>>>(hIb, WcT1, bc1, k1, nullptr, N_ITEM, 0);
    gemm_k128<<<GU, 256, 0, stream>>>(hUb, WqUT, bq_u, q1, nullptr, N_USER, 0);
    gemm_k128<<<GU, 256, 0, stream>>>(hUb, WvUT, bv_u, oriU, oriUb, N_USER, 0);
    gemm_k128<<<GI, 256, 0, stream>>>(hIb, WvIT, bv_i, oriI, oriIb, N_ITEM, 0);
    gemm_k128<<<GU, 256, 0, stream>>>(oriUb, RmsgT0, nullptr, v0, nullptr, N_USER, 0);
    gemm_k128<<<GI, 256, 0, stream>>>(oriIb, RmsgT1, nullptr, v1, nullptr, N_ITEM, 0);

    // Edge attention + dual segment softmax + scatter messages.
    const int EH = NEDGE * 8;
    const int BE = (EH + 255) / 256;
    edge_scores<<<BE, 256, 0, stream>>>(q0, k0, buys_src, buys_dst, rel_pri, raw0, rin0, EH);
    edge_scores<<<BE, 256, 0, stream>>>(q1, k1, rev_src, rev_dst, rel_pri + 8, raw1, rin1, EH);
    seg_max<<<BE, 256, 0, stream>>>(raw0, rin0, buys_dst, mlfI, mhfI, EH);
    seg_max<<<BE, 256, 0, stream>>>(raw1, rin1, rev_dst, mlfU, mhfU, EH);
    exp_sum<<<BE, 256, 0, stream>>>(raw0, rin0, buys_dst, mlfI, mhfI, zlfI, zhfI, EH);
    exp_sum<<<BE, 256, 0, stream>>>(raw1, rin1, rev_dst, mlfU, mhfU, zlfU, zhfU, EH);
    const int BA = (NEDGE * 32 + 255) / 256;
    aggregate<<<BA, 256, 0, stream>>>(v0, buys_src, buys_dst, raw0, rin0, zlfI, zhfI, ltI, htI, NEDGE);
    aggregate<<<BA, 256, 0, stream>>>(v1, rev_src, rev_dst, raw1, rin1, zlfU, zhfU, ltU, htU, NEDGE);

    // User-side update.
    prep_ltht<<<BN, 256, 0, stream>>>(oriU, ltU, htU, ltUb, htUb, NU128);
    gemm_k128<<<GU, 256, 0, stream>>>(ltUb, WfT, bF, g0b, nullptr, N_USER, 1);
    gemm_k128<<<GU, 256, 0, stream>>>(htUb, WfT, bF, g1b, nullptr, N_USER, 1);
    gemm_k128<<<GU, 256, 0, stream>>>(oriUb, WxT, bx, xpb, nullptr, N_USER, 1);
    mix_kernel<<<(N_USER + 7) / 8, 256, 0, stream>>>(g0b, g1b, xpb, ltU, htU, tb, tbb, N_USER);
    gemm_k128<<<GU, 256, 0, stream>>>(tbb, WaUT, ba_u, ub, nullptr, N_USER, 0);
    blend<<<BN, 256, 0, stream>>>(ub, h_user, skip, 0, outU, NU128);

    // Item-side update (reuses scratch).
    prep_ltht<<<BNI, 256, 0, stream>>>(oriI, ltI, htI, ltIb, htIb, NI128);
    gemm_k128<<<GI, 256, 0, stream>>>(ltIb, WfT, bF, g0b, nullptr, N_ITEM, 1);
    gemm_k128<<<GI, 256, 0, stream>>>(htIb, WfT, bF, g1b, nullptr, N_ITEM, 1);
    gemm_k128<<<GI, 256, 0, stream>>>(oriIb, WxT, bx, xpb, nullptr, N_ITEM, 1);
    mix_kernel<<<(N_ITEM + 7) / 8, 256, 0, stream>>>(g0b, g1b, xpb, ltI, htI, tb, tbb, N_ITEM);
    gemm_k128<<<GI, 256, 0, stream>>>(tbb, WaIT, ba_i, ub, nullptr, N_ITEM, 0);
    blend<<<BNI, 256, 0, stream>>>(ub, h_item, skip, 1, outI, NI128);
}
